// PointNetPlusPlusPose_89953795047896
// MI455X (gfx1250) — compile-verified, matrix operand order checked by reading, not run on hardware
//
#include <hip/hip_runtime.h>
#include <hip/hip_bf16.h>
#include <stdint.h>

// ---------------------------------------------------------------------------
// PointNet++ pose net for MI455X (gfx1250, wave32).
// GEMM path: f32 inputs pre-cast to bf16 with K padded to a multiple of 32
// (zero fill), then a WMMA GEMM (v_wmma_f32_16x16x32_bf16) runs a branch-free
// K loop; every fragment is two aligned global_load_b128s. Each wave owns a
// 16x(16*UT) output strip; UT (2 or 4) always divides N/16 exactly, so there
// is no conditional structure anywhere in the wave -> no exec juggling.
// ---------------------------------------------------------------------------

typedef __attribute__((ext_vector_type(16))) __bf16    v16bf;
typedef __attribute__((ext_vector_type(8)))  float     v8f;
typedef __attribute__((ext_vector_type(4)))  uint32_t  v4u;

#define BN_EPS 1e-5f

__device__ __forceinline__ uint32_t bfbits(float f) {
  union { float f; uint32_t u; } c; c.f = f;
  uint32_t u = c.u;
  return (u + 0x7FFFu + ((u >> 16) & 1u)) >> 16;   // round-to-nearest-even
}
__device__ __forceinline__ uint32_t pack_bf16(float lo, float hi) {
  return bfbits(lo) | (bfbits(hi) << 16);
}

// src (rows, K) f32 -> dst (rows, Kp) bf16, zero padded; one packed pair/thread
__global__ void cast_pad_bf16_kernel(const float* __restrict__ src,
                                     uint32_t* __restrict__ dst,
                                     int rows, int K, int Kp, int totalPairs)
{
  int i = blockIdx.x * blockDim.x + threadIdx.x;
  if (i >= totalPairs) return;
  int ppr = Kp >> 1;                       // pairs per row
  int r = i / ppr;
  int k = (i - r * ppr) * 2;
  float f0 = (k < K)     ? src[(size_t)r * K + k]     : 0.f;
  float f1 = (k + 1 < K) ? src[(size_t)r * K + k + 1] : 0.f;
  dst[i] = pack_bf16(f0, f1);
}

// 16-bit fragment (ISA 7.12.2): elems 0..7 = K(kb+8h .. kb+8h+7),
// elems 8..15 = K(kb+16+8h .. kb+16+8h+7)  -> two aligned b128 loads.
__device__ __forceinline__ v16bf load_frag_bf(const __bf16* __restrict__ base,
                                              int kb, int half) {
  union { v16bf v; v4u q[2]; } r;
  const int k0 = kb + 8 * half;
  r.q[0] = *(const v4u*)(base + k0);
  r.q[1] = *(const v4u*)(base + k0 + 16);
  return r.v;
}

// out[M,N] = A[M,Kp] * W[N,Kp]^T + bias[N]  (A,W bf16).
// Requires M%16==0, (N/16)%UT==0, Kp%32==0.
template <int UT>
__global__ __launch_bounds__(256) void gemm_bias_wmma(
    const __bf16* __restrict__ A, const __bf16* __restrict__ W,
    const float* __restrict__ bias, float* __restrict__ out,
    int M, int N, int Kp)
{
  const int lane = threadIdx.x & 31;
  const int wave = threadIdx.x >> 5;
  const int ntN  = N >> 4;
  const int ntM  = M >> 4;
  const int ntNg = ntN / UT;                  // exact by construction
  const int wid  = blockIdx.x * 8 + wave;     // 8 waves / block
  if (wid >= ntM * ntNg) return;              // wave-uniform exit
  const int tm = wid / ntNg, tg = wid % ntNg;
  const int row = lane & 15, half = lane >> 4;

  const __bf16* Abase = A + (size_t)(tm * 16 + row) * Kp;
  const __bf16* Bbase[UT];
#pragma unroll
  for (int u = 0; u < UT; ++u)
    Bbase[u] = W + (size_t)((tg * UT + u) * 16 + row) * Kp;

  v8f acc[UT];
#pragma unroll
  for (int u = 0; u < UT; ++u) acc[u] = (v8f){};

  for (int kb = 0; kb < Kp; kb += 32) {
    __builtin_prefetch(Abase + kb + 32, 0, 1);            // global_prefetch
    const v16bf a = load_frag_bf(Abase, kb, half);
#pragma unroll
    for (int u = 0; u < UT; ++u) {
      const v16bf b = load_frag_bf(Bbase[u], kb, half);
      acc[u] = __builtin_amdgcn_wmma_f32_16x16x32_bf16(
          false, a, false, b, (short)0, acc[u], false, false);
    }
  }

  // C/D layout: lane -> N = lane%16, VGPR i -> M = i + 8*(lane/16)
#pragma unroll
  for (int u = 0; u < UT; ++u) {
    const int n = (tg * UT + u) * 16 + row;
    const float bv = bias[n];
#pragma unroll
    for (int i = 0; i < 8; ++i) {
      const int cm = tm * 16 + i + 8 * half;
      out[(size_t)cm * N + n] = acc[u][i] + bv;
    }
  }
}

// xyz (B,3,N) -> pts (B,N,3)
__global__ void transpose_xyz_kernel(const float* __restrict__ xyz,
                                     float* __restrict__ pts, int Bb, int n)
{
  int t = blockIdx.x * blockDim.x + threadIdx.x;
  int total = Bb * n * 3;
  if (t >= total) return;
  int c = t % 3, j = (t / 3) % n, b = t / (3 * n);
  pts[t] = xyz[(size_t)b * 3 * n + (size_t)c * n + j];
}

// Farthest-point sampling: one block per batch, dist kept in registers.
__global__ __launch_bounds__(256) void fps_kernel(
    const float* __restrict__ pts, int* __restrict__ out_idx, int n, int npoint)
{
  const int b = blockIdx.x;
  const float* p = pts + (size_t)b * n * 3;
  int* oi = out_idx + (size_t)b * npoint;
  const int T = 256, tid = threadIdx.x;
  const int per = (n + T - 1) / T;            // <= 16
  float dist[16];
  for (int i = 0; i < 16; ++i) dist[i] = 1e10f;
  __shared__ float svals[256];
  __shared__ int   sidx[256];
  __shared__ float c3[3];
  int far = 0;
  for (int s = 0; s < npoint; ++s) {
    if (tid == 0) {
      oi[s] = far;
      c3[0] = p[far * 3]; c3[1] = p[far * 3 + 1]; c3[2] = p[far * 3 + 2];
    }
    __syncthreads();
    float cx = c3[0], cy = c3[1], cz = c3[2];
    float best = -1.f; int bi = 0;
    for (int i = 0; i < per; ++i) {
      int j = tid + i * T;
      if (j < n) {
        float dx = p[j * 3] - cx, dy = p[j * 3 + 1] - cy, dz = p[j * 3 + 2] - cz;
        float d = dx * dx + dy * dy + dz * dz;
        float dm = fminf(dist[i], d); dist[i] = dm;
        if (dm > best) { best = dm; bi = j; }
      }
    }
    svals[tid] = best; sidx[tid] = bi;
    __syncthreads();
    for (int off = 128; off > 0; off >>= 1) {
      if (tid < off) {
        if (svals[tid + off] > svals[tid] ||
            (svals[tid + off] == svals[tid] && sidx[tid + off] < sidx[tid])) {
          svals[tid] = svals[tid + off]; sidx[tid] = sidx[tid + off];
        }
      }
      __syncthreads();
    }
    far = sidx[0];
    __syncthreads();
  }
}

// ctr[t] = pts[b][fps_idx[t]]
__global__ void gather_centers_kernel(const float* __restrict__ pts,
                                      const int* __restrict__ fidx,
                                      float* __restrict__ ctr, int n, int S, int total)
{
  int t = blockIdx.x * blockDim.x + threadIdx.x;
  if (t >= total) return;
  int b = t / S;
  int id = fidx[t];
  const float* p = pts + ((size_t)b * n + id) * 3;
  ctr[t * 3 + 0] = p[0]; ctr[t * 3 + 1] = p[1]; ctr[t * 3 + 2] = p[2];
}

// First nsample indices (ascending) within radius; pad with first hit.
__global__ void ball_query_kernel(const float* __restrict__ pts,
                                  const float* __restrict__ ctr,
                                  int* __restrict__ idx,
                                  int n, int S, int nsample, float r2, int total)
{
  int t = blockIdx.x * blockDim.x + threadIdx.x;
  if (t >= total) return;
  int b = t / S;
  const float* p = pts + (size_t)b * n * 3;
  float cx = ctr[t * 3], cy = ctr[t * 3 + 1], cz = ctr[t * 3 + 2];
  int* o = idx + (size_t)t * nsample;
  int cnt = 0, first = -1;
  for (int j = 0; j < n && cnt < nsample; ++j) {
    float dx = p[j * 3] - cx, dy = p[j * 3 + 1] - cy, dz = p[j * 3 + 2] - cz;
    if (dx * dx + dy * dy + dz * dz <= r2) {
      if (first < 0) first = j;
      o[cnt++] = j;
    }
  }
  if (first < 0) first = 0;
  for (; cnt < nsample; ++cnt) o[cnt] = first;
}

// A[(b,s,k), 0:3] = pts[b][idx]-ctr[b,s] ; A[..., 3:3+C] = feat[b][idx]
__global__ void group_kernel(const float* __restrict__ pts,
                             const float* __restrict__ ctr,
                             const float* __restrict__ feat,
                             const int* __restrict__ idx,
                             float* __restrict__ A,
                             int n, int S, int Ks, int C, int total)
{
  int t = blockIdx.x * blockDim.x + threadIdx.x;
  if (t >= total) return;                    // total = B*S*Ks
  int k = t % Ks, bs = t / Ks, b = bs / S;
  int id = idx[(size_t)bs * Ks + k];
  int cols = 3 + C;
  float* r = A + (size_t)t * cols;
  const float* pp = pts + ((size_t)b * n + id) * 3;
  r[0] = pp[0] - ctr[bs * 3 + 0];
  r[1] = pp[1] - ctr[bs * 3 + 1];
  r[2] = pp[2] - ctr[bs * 3 + 2];
  if (C > 0) {
    const float* f = feat + ((size_t)b * n + id) * (size_t)C;
    for (int c = 0; c < C; ++c) r[3 + c] = f[c];
  }
}

// group_all: A[row] = concat(xyz[row], feat[row])
__global__ void concat_rows_kernel(const float* __restrict__ xyzr,
                                   const float* __restrict__ feat,
                                   float* __restrict__ A, int rows, int C)
{
  int t = blockIdx.x * blockDim.x + threadIdx.x;
  if (t >= rows) return;
  float* r = A + (size_t)t * (3 + C);
  r[0] = xyzr[t * 3]; r[1] = xyzr[t * 3 + 1]; r[2] = xyzr[t * 3 + 2];
  const float* f = feat + (size_t)t * C;
  for (int c = 0; c < C; ++c) r[3 + c] = f[c];
}

// per-channel mean/var over all rows -> mv[0:C]=mean, mv[C:2C]=var
__global__ __launch_bounds__(256) void channel_meanvar_kernel(
    const float* __restrict__ x, float* __restrict__ mv, int Mrows, int C)
{
  int ch = blockIdx.x, tid = threadIdx.x;
  __shared__ float s1[256], s2[256];
  float a = 0.f, b = 0.f;
  for (int r = tid; r < Mrows; r += 256) {
    float v = x[(size_t)r * C + ch];
    a += v; b += v * v;
  }
  s1[tid] = a; s2[tid] = b;
  __syncthreads();
  for (int off = 128; off > 0; off >>= 1) {
    if (tid < off) { s1[tid] += s1[tid + off]; s2[tid] += s2[tid + off]; }
    __syncthreads();
  }
  if (tid == 0) {
    float m = s1[0] / (float)Mrows;
    mv[ch] = m;
    mv[C + ch] = s2[0] / (float)Mrows - m * m;
  }
}

__global__ void bn_relu_kernel(float* __restrict__ x, const float* __restrict__ mv,
                               const float* __restrict__ g, const float* __restrict__ beta,
                               size_t total, int C)
{
  size_t i = (size_t)blockIdx.x * blockDim.x + threadIdx.x;
  if (i >= total) return;
  int ch = (int)(i % C);
  float m = mv[ch], v = mv[C + ch];
  float val = (x[i] - m) * rsqrtf(v + BN_EPS) * g[ch] + beta[ch];
  x[i] = fmaxf(val, 0.f);
}

// (BS, Ks, C) -> (BS, C) max over Ks
__global__ void maxpool_kernel(const float* __restrict__ x, float* __restrict__ out,
                               int BS, int Ks, int C)
{
  int t = blockIdx.x * blockDim.x + threadIdx.x;
  if (t >= BS * C) return;
  int ch = t % C, bs = t / C;
  float m = -3.4e38f;
  const float* p = x + (size_t)bs * Ks * C + ch;
  for (int k = 0; k < Ks; ++k) m = fmaxf(m, p[(size_t)k * C]);
  out[(size_t)bs * C + ch] = m;
}

// decoder BN over the 16-row batch, fused stats+normalize+relu (one thread/channel)
__global__ void bn_relu_rows_kernel(float* __restrict__ x,
                                    const float* __restrict__ g,
                                    const float* __restrict__ beta,
                                    int Mrows, int C)
{
  int ch = blockIdx.x * blockDim.x + threadIdx.x;
  if (ch >= C) return;
  float s = 0.f, s2 = 0.f;
  for (int r = 0; r < Mrows; ++r) {
    float v = x[(size_t)r * C + ch];
    s += v; s2 += v * v;
  }
  float m = s / (float)Mrows;
  float var = s2 / (float)Mrows - m * m;
  float sc = rsqrtf(var + BN_EPS) * g[ch];
  float bt = beta[ch];
  for (int r = 0; r < Mrows; ++r) {
    float v = (x[(size_t)r * C + ch] - m) * sc + bt;
    x[(size_t)r * C + ch] = fmaxf(v, 0.f);
  }
}

// t = y@wt^T+bt (out[0:48]), q = y@wq^T+bq (out[48:96])
__global__ void heads_kernel(const float* __restrict__ y,
                             const float* __restrict__ wt, const float* __restrict__ bt,
                             const float* __restrict__ wq, const float* __restrict__ bq,
                             float* __restrict__ out)
{
  int t = blockIdx.x * blockDim.x + threadIdx.x;
  if (t >= 16 * 6) return;
  int b = t / 6, j = t % 6, head = j / 3, o = j % 3;
  const float* w  = head ? wq : wt;
  const float* bb = head ? bq : bt;
  float s = bb[o];
  const float* yr = y + (size_t)b * 1024;
  const float* wr = w + (size_t)o * 1024;
  for (int c = 0; c < 1024; ++c) s += yr[c] * wr[c];
  out[head * 48 + b * 3 + o] = s;
}

// ---------------------------------------------------------------------------

extern "C" void kernel_launch(void* const* d_in, const int* in_sizes, int n_in,
                              void* d_out, int out_size, void* d_ws, size_t ws_size,
                              hipStream_t stream) {
  (void)in_sizes; (void)n_in; (void)out_size; (void)ws_size;
  const int Bb = 16, Np = 4096;
  const int S1 = 512, K1 = 32;
  const int S2 = 128, K2 = 16;

  // Inputs in setup_inputs() insertion order:
  // 0:xyz; sa1: w0..2, b0..2, g0..2, beta0..2 (1..12); sa2 (13..24); sa3 (25..36);
  // dec: w0..3 (37..40), b0..3 (41..44), g0..3 (45..48), beta0..3 (49..52);
  // 53:wt 54:bt 55:wq 56:bq
  auto F = [&](int i) { return (const float*)d_in[i]; };
  const float* xyz = F(0);

  float* ws = (float*)d_ws;
  size_t o = 0;
  auto alloc = [&](size_t nf) { float* p = ws + o; o += (nf + 63) & ~(size_t)63; return p; };
  float* pts1    = alloc((size_t)Bb * Np * 3);        // (B, N, 3)
  float* l1x     = alloc((size_t)Bb * S1 * 3);
  float* l1p     = alloc((size_t)Bb * S1 * 64);
  float* l2x     = alloc((size_t)Bb * S2 * 3);
  float* l2p     = alloc((size_t)Bb * S2 * 256);
  float* y0      = alloc((size_t)Bb * 1024);
  float* y1      = alloc((size_t)Bb * 1024);
  float* mv      = alloc(2048);
  int*   fps_idx = (int*)alloc((size_t)Bb * S1);
  int*   bq_idx  = (int*)alloc((size_t)Bb * S1 * K1);
  float* bufA    = alloc((size_t)Bb * S1 * K1 * 64);    // 16.7M f
  float* bufB    = alloc((size_t)Bb * S1 * K1 * 32);    // 8.4M f
  float* bufC    = alloc((size_t)Bb * S2 * K2 * 67);    // grouped inputs (max)
  uint32_t* abf  = (uint32_t*)alloc((size_t)Bb * S1 * K1 * 32 / 2); // bf16 A, 8.4M elems
  uint32_t* wbf  = (uint32_t*)alloc((size_t)1024 * 1024 / 2);       // bf16 W, 1.05M elems

  auto blk = [](size_t tot) { return (int)((tot + 255) / 256); };

  // cast+pad f32 -> bf16 (row stride Kp), then WMMA GEMM (UT N-tiles per wave)
  auto launch_gemm = [&](const float* Ain, const float* w, const float* bias,
                         float* Out, int M, int Nc, int Kc) {
    const int Kp = (Kc + 31) & ~31;
    int pa = M * (Kp / 2);
    cast_pad_bf16_kernel<<<blk(pa), 256, 0, stream>>>(Ain, abf, M, Kc, Kp, pa);
    int pw = Nc * (Kp / 2);
    cast_pad_bf16_kernel<<<blk(pw), 256, 0, stream>>>(w, wbf, Nc, Kc, Kp, pw);
    const int ntN = Nc / 16;
    if ((ntN & 3) == 0) {
      int tiles = (M / 16) * (ntN / 4);
      gemm_bias_wmma<4><<<(tiles + 7) / 8, 256, 0, stream>>>(
          (const __bf16*)abf, (const __bf16*)wbf, bias, Out, M, Nc, Kp);
    } else {                                   // N == 32 layers
      int tiles = (M / 16) * (ntN / 2);
      gemm_bias_wmma<2><<<(tiles + 7) / 8, 256, 0, stream>>>(
          (const __bf16*)abf, (const __bf16*)wbf, bias, Out, M, Nc, Kp);
    }
  };

  // GEMM + global BN + ReLU (one MLP layer)
  auto run_layer = [&](const float* Ain, float* Out, const float* w, const float* bias,
                       const float* g, const float* beta, int M, int Kc, int Nc) {
    launch_gemm(Ain, w, bias, Out, M, Nc, Kc);
    channel_meanvar_kernel<<<Nc, 256, 0, stream>>>(Out, mv, M, Nc);
    size_t tot = (size_t)M * Nc;
    bn_relu_kernel<<<blk(tot), 256, 0, stream>>>(Out, mv, g, beta, tot, Nc);
  };

  // ---- transpose (B,3,N) -> (B,N,3)
  transpose_xyz_kernel<<<blk((size_t)Bb * Np * 3), 256, 0, stream>>>(xyz, pts1, Bb, Np);

  // ---- SA1: fps 512, ball query r=4 k=32, MLP 3->32->32->64, maxpool
  fps_kernel<<<Bb, 256, 0, stream>>>(pts1, fps_idx, Np, S1);
  gather_centers_kernel<<<blk(Bb * S1), 256, 0, stream>>>(pts1, fps_idx, l1x, Np, S1, Bb * S1);
  ball_query_kernel<<<blk(Bb * S1), 256, 0, stream>>>(pts1, l1x, bq_idx, Np, S1, K1, 16.0f, Bb * S1);
  {
    size_t tot = (size_t)Bb * S1 * K1;
    group_kernel<<<blk(tot), 256, 0, stream>>>(pts1, l1x, nullptr, bq_idx, bufC,
                                               Np, S1, K1, 0, (int)tot);
    int M = (int)tot;
    run_layer(bufC, bufA, F(1), F(4),  F(7),  F(10), M, 3, 32);
    run_layer(bufA, bufB, F(2), F(5),  F(8),  F(11), M, 32, 32);
    run_layer(bufB, bufA, F(3), F(6),  F(9),  F(12), M, 32, 64);
    maxpool_kernel<<<blk((size_t)Bb * S1 * 64), 256, 0, stream>>>(bufA, l1p, Bb * S1, K1, 64);
  }

  // ---- SA2: fps 128 (over l1x), ball query r=8 k=16, MLP 67->64->128->256
  fps_kernel<<<Bb, 256, 0, stream>>>(l1x, fps_idx, S1, S2);
  gather_centers_kernel<<<blk(Bb * S2), 256, 0, stream>>>(l1x, fps_idx, l2x, S1, S2, Bb * S2);
  ball_query_kernel<<<blk(Bb * S2), 256, 0, stream>>>(l1x, l2x, bq_idx, S1, S2, K2, 64.0f, Bb * S2);
  {
    size_t tot = (size_t)Bb * S2 * K2;
    group_kernel<<<blk(tot), 256, 0, stream>>>(l1x, l2x, l1p, bq_idx, bufC,
                                               S1, S2, K2, 64, (int)tot);
    int M = (int)tot;
    run_layer(bufC, bufA, F(13), F(16), F(19), F(22), M, 67, 64);
    run_layer(bufA, bufB, F(14), F(17), F(20), F(23), M, 64, 128);
    run_layer(bufB, bufA, F(15), F(18), F(21), F(24), M, 128, 256);
    maxpool_kernel<<<blk((size_t)Bb * S2 * 256), 256, 0, stream>>>(bufA, l2p, Bb * S2, K2, 256);
  }

  // ---- SA3 (group all): concat(l2x, l2p) -> MLP 259->256->512->1024, maxpool over 128
  {
    int rows = Bb * S2;                                 // 2048
    concat_rows_kernel<<<blk(rows), 256, 0, stream>>>(l2x, l2p, bufC, rows, 256);
    run_layer(bufC, bufA, F(25), F(28), F(31), F(34), rows, 259, 256);
    run_layer(bufA, bufB, F(26), F(29), F(32), F(35), rows, 256, 512);
    run_layer(bufB, bufA, F(27), F(30), F(33), F(36), rows, 512, 1024);
    maxpool_kernel<<<blk((size_t)Bb * 1024), 256, 0, stream>>>(bufA, y0, Bb, S2, 1024);
  }

  // ---- decoder: 4 x (GEMM 16x1024x1024 + BN-over-batch + ReLU), ping-pong y0/y1
  {
    float* cur = y0; float* nxt = y1;
    for (int l = 0; l < 4; ++l) {
      launch_gemm(cur, F(37 + l), F(41 + l), nxt, Bb, 1024, 1024);
      bn_relu_rows_kernel<<<blk(1024), 256, 0, stream>>>(nxt, F(45 + l), F(49 + l), Bb, 1024);
      float* tmp = cur; cur = nxt; nxt = tmp;
    }
    // after 4 swaps result is back in y0
    heads_kernel<<<1, 128, 0, stream>>>(y0, F(53), F(54), F(55), F(56), (float*)d_out);
  }
}
